// SystemCausalReasoner_62749472194562
// MI455X (gfx1250) — matrix operand order repaired and checked by reading, not compiled, not run on hardware
//
#include <hip/hip_runtime.h>
#include <hip/hip_bf16.h>

typedef __bf16 bf16_t;
typedef __attribute__((ext_vector_type(16))) __bf16 v16bf;
typedef __attribute__((ext_vector_type(8)))  __bf16 v8bf;
typedef __attribute__((ext_vector_type(8)))  float  v8f;
typedef __attribute__((ext_vector_type(4)))  unsigned int v4u;
typedef __attribute__((ext_vector_type(8)))  int v8i;
typedef __attribute__((ext_vector_type(4)))  int v4i;

#define DD   2048
#define BB   32768
#define LN_EPS 1e-5f

#if defined(__has_builtin)
#if __has_builtin(__builtin_amdgcn_tensor_load_to_lds)
#define HAVE_TDM 1
#else
#define HAVE_TDM 0
#endif
#else
#define HAVE_TDM 0
#endif

// ---------------------------------------------------------------------------
// f32 -> bf16 conversion (grid-stride)
// ---------------------------------------------------------------------------
__global__ __launch_bounds__(256) void cvt_f32_to_bf16(const float* __restrict__ src,
                                                       bf16_t* __restrict__ dst,
                                                       long long n) {
    long long i = (long long)blockIdx.x * blockDim.x + threadIdx.x;
    long long stride = (long long)gridDim.x * blockDim.x;
    for (; i < n; i += stride) dst[i] = (bf16_t)src[i];
}

// ---------------------------------------------------------------------------
// Tiled bf16 GEMM with f32 WMMA accumulate:
//   out[M,N] = A[M,K] @ W[N,K]^T + bias  (+ resid) (ReLU?) -> bf16 or f32
// Block tile 128x128, BK=64, 256 threads = 8 waves (2x4), wave tile 64x32.
// A tile: vmem->LDS (padded stride 72).  W tile: TDM tensor_load_to_lds with
// descriptor padding (32 DWORDs interval, 4 DWORDs pad) -> same 72 stride.
// ---------------------------------------------------------------------------
template <bool RESID, bool RELU, bool OUT_BF16>
__global__ __launch_bounds__(256)
void gemm_bf16_wmma(const bf16_t* __restrict__ A, int lda,
                    const bf16_t* __restrict__ W,            // N x K, ldw == K
                    const float*  __restrict__ bias,
                    const bf16_t* __restrict__ resid, int ldres,
                    void* __restrict__ outp, int ldout, int K) {
    constexpr int LDT = 72;                       // padded LDS row stride (bf16)
    __shared__ bf16_t sA[128 * LDT];
    __shared__ bf16_t sW[128 * LDT];

    const int tid  = threadIdx.x;
    const int lane = tid & 31;
    const int wave = tid >> 5;
    const int wm   = wave >> 2;                   // 0..1  -> 64 rows each
    const int wn   = wave & 3;                    // 0..3  -> 32 cols each
    const int half = lane >> 4;                   // 0/1
    const int l15  = lane & 15;

    const int bm = blockIdx.y * 128;
    const int bn = blockIdx.x * 128;

    const v8f zv = {0.f, 0.f, 0.f, 0.f, 0.f, 0.f, 0.f, 0.f};
    v8f acc[4][2];
#pragma unroll
    for (int ta = 0; ta < 4; ++ta)
#pragma unroll
        for (int tb = 0; tb < 2; ++tb) acc[ta][tb] = zv;

#if HAVE_TDM
    const unsigned ldsW = (unsigned)(size_t)(const void*)sW;
#endif

    for (int k0 = 0; k0 < K; k0 += 64) {
        __syncthreads();                          // prior compute done

#if HAVE_TDM
        if (tid < 32) {
            // ---- Tensor Data Mover: DMA W tile (128 rows x 64 bf16) -> sW ----
            unsigned long long ga =
                (unsigned long long)(size_t)(const void*)(W + (size_t)bn * K + k0);
            v4u g0;
            g0.x = 1u;                                            // count=1
            g0.y = ldsW;                                          // lds_addr (bytes)
            g0.z = (unsigned)(ga & 0xffffffffu);                  // global_addr lo
            g0.w = (unsigned)((ga >> 32) & 0x01ffffffu) | (2u << 30); // hi | type=2
            v8i g1;
            // data_size=2B(code1), pad_enable, pad_interval=32DW(code4),
            // pad_amount=4DW(code3): 64 bf16 row + 16B pad = 72-elem stride
            g1[0] = (1 << 16) | (1 << 20) | (4 << 22) | (3 << 25);
            g1[1] = (int)(((unsigned)K & 0xffffu) << 16);         // tensor_dim0 lo16
            g1[2] = (int)(((unsigned)K >> 16) | (128u << 16));    // dim0 hi | dim1=128
            g1[3] = (int)(64u << 16);                             // tile_dim0=64
            g1[4] = 128;                                          // tile_dim1=128
            g1[5] = (int)(unsigned)K;                             // dim0_stride lo32
            g1[6] = 0;
            g1[7] = 0;
            v4i gz = {0, 0, 0, 0};
#if __clang_major__ >= 23
            v8i gz8 = {0, 0, 0, 0, 0, 0, 0, 0};
            __builtin_amdgcn_tensor_load_to_lds(g0, g1, gz, gz, gz8, 0);
#else
            __builtin_amdgcn_tensor_load_to_lds(g0, g1, gz, gz, 0);
#endif
        }
#endif

        // ---- cooperative A fill: 128 rows x 64 bf16, 16B chunks ----
#pragma unroll
        for (int c = 0; c < 4; ++c) {
            int idx = tid + c * 256;              // 0..1023
            int row = idx >> 3;                   // 0..127
            int col = (idx & 7) << 3;             // 0,8,...,56
            *(v8bf*)&sA[row * LDT + col] =
                *(const v8bf*)(A + (size_t)(bm + row) * lda + k0 + col);
#if !HAVE_TDM
            *(v8bf*)&sW[row * LDT + col] =
                *(const v8bf*)(W + (size_t)(bn + row) * K + k0 + col);
#endif
        }
        // prefetch next A K-slice into L2 (global_prefetch_b8)
        if (k0 + 64 < K) {
            int pr = tid >> 1;
            int pc = (tid & 1) << 5;
            __builtin_prefetch(A + (size_t)(bm + pr) * lda + k0 + 64 + pc, 0, 1);
        }
#if HAVE_TDM
        if (tid < 32) __builtin_amdgcn_s_wait_tensorcnt(0);
#endif
        __syncthreads();

#pragma unroll
        for (int kk = 0; kk < 2; ++kk) {          // two 16x16x32 k-steps
            v16bf afrag[4], bfrag[2];
#pragma unroll
            for (int ta = 0; ta < 4; ++ta) {
                // A 16x32 bf16 layout: lane row = l15; K chunks [h*8,h*8+8) and +16
                const bf16_t* p = &sA[(wm * 64 + ta * 16 + l15) * LDT + kk * 32 + half * 8];
                union { v16bf v; v8bf h[2]; } u;
                u.h[0] = *(const v8bf*)p;
                u.h[1] = *(const v8bf*)(p + 16);
                afrag[ta] = u.v;
            }
#pragma unroll
            for (int tb = 0; tb < 2; ++tb) {
                // B 32x16 bf16 layout: lane col = l15; K chunks [h*8,h*8+8) and +16
                const bf16_t* p = &sW[(wn * 32 + tb * 16 + l15) * LDT + kk * 32 + half * 8];
                union { v16bf v; v8bf h[2]; } u;
                u.h[0] = *(const v8bf*)p;
                u.h[1] = *(const v8bf*)(p + 16);
                bfrag[tb] = u.v;
            }
#pragma unroll
            for (int ta = 0; ta < 4; ++ta)
#pragma unroll
                for (int tb = 0; tb < 2; ++tb)
                    acc[ta][tb] = __builtin_amdgcn_wmma_f32_16x16x32_bf16(
                        false, afrag[ta], false, bfrag[tb],
                        (short)0, acc[ta][tb], false, false);
        }
    }

    // epilogue: C layout -> lane holds column (l15), rows half*8 + r
#pragma unroll
    for (int ta = 0; ta < 4; ++ta) {
#pragma unroll
        for (int tb = 0; tb < 2; ++tb) {
            const int n  = bn + wn * 32 + tb * 16 + l15;
            const int mb = bm + wm * 64 + ta * 16 + half * 8;
            const float bv = bias[n];
#pragma unroll
            for (int r = 0; r < 8; ++r) {
                float v = acc[ta][tb][r] + bv;
                if (RESID) v += (float)resid[(size_t)(mb + r) * ldres + n];
                if (RELU)  v = fmaxf(v, 0.f);
                if (OUT_BF16)
                    ((bf16_t*)outp)[(size_t)(mb + r) * ldout + n] = (bf16_t)v;
                else
                    ((float*)outp)[(size_t)(mb + r) * ldout + n] = v;
            }
        }
    }
}

// ---------------------------------------------------------------------------
// Row LayerNorm over D=2048 (f32 in, bf16 out into strided slice)
// ---------------------------------------------------------------------------
__global__ __launch_bounds__(256)
void layernorm_rows(const float* __restrict__ x,
                    const float* __restrict__ g,
                    const float* __restrict__ b,
                    bf16_t* __restrict__ dst, int ldst) {
    __shared__ float s1[256];
    __shared__ float s2[256];
    const int tid = threadIdx.x;
    const float* xr = x + (size_t)blockIdx.x * DD;

    float xv[8];
    float sum = 0.f, sq = 0.f;
#pragma unroll
    for (int j = 0; j < 8; ++j) {
        float v = xr[tid + j * 256];
        xv[j] = v;
        sum += v;
        sq  += v * v;
    }
    s1[tid] = sum;
    s2[tid] = sq;
    __syncthreads();
    for (int off = 128; off > 0; off >>= 1) {
        if (tid < off) { s1[tid] += s1[tid + off]; s2[tid] += s2[tid + off]; }
        __syncthreads();
    }
    const float mu  = s1[0] * (1.f / (float)DD);
    const float var = s2[0] * (1.f / (float)DD) - mu * mu;
    const float rs  = rsqrtf(var + LN_EPS);

    bf16_t* dr = dst + (size_t)blockIdx.x * ldst;
#pragma unroll
    for (int j = 0; j < 8; ++j) {
        int col = tid + j * 256;
        float y = (xv[j] - mu) * rs * g[col] + b[col];
        dr[col] = (bf16_t)y;
    }
}

// ---------------------------------------------------------------------------
// Host orchestration
// ---------------------------------------------------------------------------
static inline size_t align_up(size_t v, size_t a) { return (v + a - 1) & ~(a - 1); }

extern "C" void kernel_launch(void* const* d_in, const int* in_sizes, int n_in,
                              void* d_out, int out_size, void* d_ws, size_t ws_size,
                              hipStream_t stream) {
    (void)in_sizes; (void)n_in; (void)out_size; (void)ws_size;
    const int D = DD, B = BB;

    // ---- inputs (setup_inputs dict order) ----
    const float* z_global   = (const float*)d_in[0];
    const float* state_w    = (const float*)d_in[1];
    const float* state_b    = (const float*)d_in[2];
    const float* behavior_w = (const float*)d_in[3];
    const float* behavior_b = (const float*)d_in[4];
    const float* path_w     = (const float*)d_in[5];
    const float* path_b     = (const float*)d_in[6];
    const float* sb_in_w  = (const float*)d_in[7];
    const float* sb_in_b  = (const float*)d_in[8];
    const float* sb_out_w = (const float*)d_in[9];
    const float* sb_out_b = (const float*)d_in[10];
    const float* bp_in_w  = (const float*)d_in[11];
    const float* bp_in_b  = (const float*)d_in[12];
    const float* bp_out_w = (const float*)d_in[13];
    const float* bp_out_b = (const float*)d_in[14];
    const float* ps_in_w  = (const float*)d_in[15];
    const float* ps_in_b  = (const float*)d_in[16];
    const float* ps_out_w = (const float*)d_in[17];
    const float* ps_out_b = (const float*)d_in[18];
    const float* ln_s_g   = (const float*)d_in[19];
    const float* ln_s_b   = (const float*)d_in[20];
    const float* ln_b_g   = (const float*)d_in[21];
    const float* ln_b_b   = (const float*)d_in[22];
    const float* ln_p_g   = (const float*)d_in[23];
    const float* ln_p_b   = (const float*)d_in[24];
    const float* fuse_w1  = (const float*)d_in[25];
    const float* fuse_b1  = (const float*)d_in[26];
    const float* fuse_w2  = (const float*)d_in[27];
    const float* fuse_b2  = (const float*)d_in[28];

    // ---- workspace carving ----
    char* ws = (char*)d_ws;
    size_t off = 0;
    auto carve = [&](size_t bytes) -> char* {
        char* p = ws + off;
        off = align_up(off + bytes, 256);
        return p;
    };
    bf16_t* zg   = (bf16_t*)carve((size_t)B * D * 2);          // z_global bf16
    bf16_t* zbuf = (bf16_t*)carve((size_t)B * 3 * D * 2);      // [z_s | z_b | z_p]
    bf16_t* vbuf = (bf16_t*)carve((size_t)B * D * 2);          // mha value tmp
    char*   preh = carve((size_t)B * D * 4);                   // pre-LN f32 / h bf16 (aliased)
    float*  pre  = (float*)preh;
    bf16_t* hbuf = (bf16_t*)preh;                              // B x 2D bf16 == same bytes

    bf16_t* w_state = (bf16_t*)carve((size_t)D * D * 2);
    bf16_t* w_behav = (bf16_t*)carve((size_t)D * D * 2);
    bf16_t* w_path  = (bf16_t*)carve((size_t)D * D * 2);
    bf16_t* wv_sb   = (bf16_t*)carve((size_t)D * D * 2);
    bf16_t* wv_bp   = (bf16_t*)carve((size_t)D * D * 2);
    bf16_t* wv_ps   = (bf16_t*)carve((size_t)D * D * 2);
    bf16_t* wo_sb   = (bf16_t*)carve((size_t)D * D * 2);
    bf16_t* wo_bp   = (bf16_t*)carve((size_t)D * D * 2);
    bf16_t* wo_ps   = (bf16_t*)carve((size_t)D * D * 2);
    bf16_t* fw1     = (bf16_t*)carve((size_t)2 * D * 3 * D * 2);
    bf16_t* fw2     = (bf16_t*)carve((size_t)D * 2 * D * 2);

    // ---- convert weights + input to bf16 ----
    auto cvt = [&](const float* s, bf16_t* d, long long n) {
        cvt_f32_to_bf16<<<4096, 256, 0, stream>>>(s, d, n);
    };
    const long long DD2 = (long long)D * D;
    cvt(z_global, zg, (long long)B * D);
    cvt(state_w, w_state, DD2);
    cvt(behavior_w, w_behav, DD2);
    cvt(path_w, w_path, DD2);
    cvt(sb_in_w + (size_t)2 * D * D, wv_sb, DD2);   // Wv = in_w[2D:]
    cvt(bp_in_w + (size_t)2 * D * D, wv_bp, DD2);
    cvt(ps_in_w + (size_t)2 * D * D, wv_ps, DD2);
    cvt(sb_out_w, wo_sb, DD2);
    cvt(bp_out_w, wo_bp, DD2);
    cvt(ps_out_w, wo_ps, DD2);
    cvt(fuse_w1, fw1, (long long)2 * D * 3 * D);
    cvt(fuse_w2, fw2, (long long)D * 2 * D);

    const dim3 blk(256);
    const dim3 gD(D / 128, B / 128);        // N = 2048
    const dim3 g2D(2 * D / 128, B / 128);   // N = 4096

    bf16_t* zs = zbuf;                      // column slices, row stride 3D
    bf16_t* zb = zbuf + D;
    bf16_t* zp = zbuf + 2 * D;
    const int ld3 = 3 * D;

    // ---- input projections: z_x = zg @ W.T + b  -> bf16 slices ----
    gemm_bf16_wmma<false, false, true><<<gD, blk, 0, stream>>>(
        zg, D, w_state, state_b, (const bf16_t*)nullptr, 0, zs, ld3, D);
    gemm_bf16_wmma<false, false, true><<<gD, blk, 0, stream>>>(
        zg, D, w_behav, behavior_b, (const bf16_t*)nullptr, 0, zb, ld3, D);
    gemm_bf16_wmma<false, false, true><<<gD, blk, 0, stream>>>(
        zg, D, w_path, path_b, (const bf16_t*)nullptr, 0, zp, ld3, D);

    // ---- DEPTH iterations of the cyclic mha+LN blocks ----
    for (int it = 0; it < 2; ++it) {
        // z_s = LN(z_s + mha(z_b; sb))
        gemm_bf16_wmma<false, false, true><<<gD, blk, 0, stream>>>(
            zb, ld3, wv_sb, sb_in_b + 2 * D, (const bf16_t*)nullptr, 0, vbuf, D, D);
        gemm_bf16_wmma<true, false, false><<<gD, blk, 0, stream>>>(
            vbuf, D, wo_sb, sb_out_b, zs, ld3, pre, D, D);
        layernorm_rows<<<B, blk, 0, stream>>>(pre, ln_s_g, ln_s_b, zs, ld3);

        // z_b = LN(z_b + mha(z_p; bp))
        gemm_bf16_wmma<false, false, true><<<gD, blk, 0, stream>>>(
            zp, ld3, wv_bp, bp_in_b + 2 * D, (const bf16_t*)nullptr, 0, vbuf, D, D);
        gemm_bf16_wmma<true, false, false><<<gD, blk, 0, stream>>>(
            vbuf, D, wo_bp, bp_out_b, zb, ld3, pre, D, D);
        layernorm_rows<<<B, blk, 0, stream>>>(pre, ln_b_g, ln_b_b, zb, ld3);

        // z_p = LN(z_p + mha(z_s; ps))   (uses the freshly updated z_s)
        gemm_bf16_wmma<false, false, true><<<gD, blk, 0, stream>>>(
            zs, ld3, wv_ps, ps_in_b + 2 * D, (const bf16_t*)nullptr, 0, vbuf, D, D);
        gemm_bf16_wmma<true, false, false><<<gD, blk, 0, stream>>>(
            vbuf, D, wo_ps, ps_out_b, zp, ld3, pre, D, D);
        layernorm_rows<<<B, blk, 0, stream>>>(pre, ln_p_g, ln_p_b, zp, ld3);
    }

    // ---- fuse: h = relu(concat @ fuse_w1.T + b1)   (concat is zbuf, K = 3D) ----
    gemm_bf16_wmma<false, true, true><<<g2D, blk, 0, stream>>>(
        zbuf, ld3, fw1, fuse_b1, (const bf16_t*)nullptr, 0, hbuf, 2 * D, 3 * D);

    // ---- out = h @ fuse_w2.T + b2 -> f32 d_out ----
    gemm_bf16_wmma<false, false, false><<<gD, blk, 0, stream>>>(
        hbuf, 2 * D, fw2, fuse_b2, (const bf16_t*)nullptr, 0, d_out, D, 2 * D);
}
